// SeperateInstMap_3590592659707
// MI455X (gfx1250) — compile-verified, matrix operand order checked by reading, not run on hardware
//
#include <hip/hip_runtime.h>

static constexpr int H = 1024, W = 2048;
static constexpr int HW = H * W;                 // 2,097,152
static constexpr int NINST = 128;
static constexpr int NCLS  = 10;                 // only classes 24..33 matter
static constexpr int NTAB  = (NINST + 1) * NCLS; // 1290 table entries

typedef __attribute__((ext_vector_type(4)))  float    f32x4;
typedef __attribute__((ext_vector_type(4)))  int      i32x4;
typedef __attribute__((ext_vector_type(16))) _Float16 v16h;
typedef __attribute__((ext_vector_type(8)))  float    v8f;

// ---------------- kernel 0: zero the global count/sum tables -----------------
__global__ void k0_zero(unsigned* __restrict__ gc, float* __restrict__ gs) {
  int t = blockIdx.x * blockDim.x + threadIdx.x;
  if (t < NTAB) { gc[t] = 0u; gs[t] = 0.0f; }
}

// ---------------- kernel 1: fused histogram + prob-sums + mask write ---------
// 512 blocks x 256 threads; block b owns quads [b*1024, (b+1)*1024).
__global__ void __launch_bounds__(256) k1_hist_masks(
    const int*   __restrict__ seg,
    const int*   __restrict__ im,
    const float* __restrict__ probs,   // (34, H, W); we touch planes 24..33 only
    float*       __restrict__ masks,   // d_out head: (128, H, W) as float
    unsigned*    __restrict__ gc,
    float*       __restrict__ gs)
{
  __shared__ unsigned lc[NTAB];
  __shared__ float    ls[NTAB];
  for (int e = threadIdx.x; e < NTAB; e += 256) { lc[e] = 0u; ls[e] = 0.0f; }
  __syncthreads();

  const int qbase = blockIdx.x * 1024 + threadIdx.x;

  for (int k = 0; k < 4; ++k) {
    const int q = qbase + k * 256;
    const int p = q * 4;                       // first pixel of this quad

    i32x4 s4 = *(const i32x4*)(seg + p);
    i32x4 m4 = *(const i32x4*)(im  + p);

    int ie[4];
#pragma unroll
    for (int j = 0; j < 4; ++j) {
      int  sv    = s4[j];
      bool thing = (sv >= 24) & (sv <= 33);    // (seg > 23.99) & (seg <= 33)
      int  iv    = thing ? m4[j] : 0;
      ie[j] = iv;
      if (iv > 0) atomicAdd(&lc[iv * NCLS + (sv - 24)], 1u);
    }

    if ((ie[0] | ie[1] | ie[2] | ie[3]) != 0) {
#pragma unroll
      for (int c = 0; c < NCLS; ++c) {
        f32x4 pr = *(const f32x4*)(probs + (size_t)(24 + c) * HW + p);
#pragma unroll
        for (int j = 0; j < 4; ++j)
          if (ie[j] > 0) atomicAdd(&ls[ie[j] * NCLS + c], pr[j]);
      }
    }

    // Mask slice: stream non-temporal zeros (1 GB, no reuse), then overwrite
    // the <=4 one-positions. Same-wave stores to one address stay ordered.
    f32x4 z = {0.0f, 0.0f, 0.0f, 0.0f};
    float* op = masks + p;
#pragma unroll 4
    for (int i = 0; i < NINST; ++i) {
      __builtin_nontemporal_store(z, (f32x4*)op);
      op += HW;
    }
#pragma unroll
    for (int j = 0; j < 4; ++j)
      if (ie[j] > 0) masks[(size_t)(ie[j] - 1) * HW + p + j] = 1.0f;
  }

  __syncthreads();
  for (int e = threadIdx.x; e < NTAB; e += 256) {
    unsigned cv = lc[e];
    if (cv)          atomicAdd(gc + e, cv);
    float sv = ls[e];
    if (sv != 0.0f)  atomicAdd(gs + e, sv);
  }
}

// ---------------- kernel 2: finalize (WMMA row-reduction + argmax) -----------
__global__ void __launch_bounds__(256) k2_finalize(
    const unsigned* __restrict__ gc,
    const float*    __restrict__ gs,
    const float*    __restrict__ iprobs,
    float*          __restrict__ out)
{
  __shared__ float totals[NINST];

  const int tid  = threadIdx.x;
  const int wave = tid >> 5;          // 8 waves, wave t owns instances 16t..16t+15
  const int lane = tid & 31;

  {
    const int row = 16 * wave + (lane & 15);            // instance index 0..127
    const unsigned* cr = gc + (row + 1) * NCLS;         // table row (skip inst 0)

    v16h a, b;
#pragma unroll
    for (int e = 0; e < 16; ++e) { a[e] = (_Float16)0.0f; b[e] = (_Float16)1.0f; }
    // A(16x32) f16 layout: lanes 0-15 hold row M=L, K{0..7,16..23};
    // lanes 16-31 hold row M=L-16, K{8..15,24..31}. B==1 everywhere, so
    // D[m,n] = sum_k A[m,k] = total[m] regardless of per-slot K placement.
    if (lane < 16) {
#pragma unroll
      for (int e = 0; e < 8; ++e) a[e] = (_Float16)(float)cr[e];   // K = 0..7
    } else {
      a[0] = (_Float16)(float)cr[8];                               // K = 8
      a[1] = (_Float16)(float)cr[9];                               // K = 9
    }
    v8f d;
#pragma unroll
    for (int e = 0; e < 8; ++e) d[e] = 0.0f;
    d = __builtin_amdgcn_wmma_f32_16x16x32_f16(
        /*neg_a=*/false, a, /*neg_b=*/false, b,
        /*c_mod=*/(short)0, d, /*reuse_a=*/false, /*reuse_b=*/false);
    // D layout: VGPR r -> M=r (lanes 0-15) / M=8+r (lanes 16-31), N = lane%16.
    if (lane == 0) {
#pragma unroll
      for (int r = 0; r < 8; ++r) totals[16 * wave + r] = d[r];
    }
    if (lane == 16) {
#pragma unroll
      for (int r = 0; r < 8; ++r) totals[16 * wave + 8 + r] = d[r];
    }
  }
  __syncthreads();

  if (tid < NINST) {
    const int i = tid;
    const unsigned* cr = gc + (i + 1) * NCLS;
    const float*    sr = gs + (i + 1) * NCLS;

    unsigned best = 0u, itot = 0u;
    int bj = -1;
#pragma unroll
    for (int j = 0; j < NCLS; ++j) {       // strict '>' == jnp.argmax first-max
      unsigned cv = cr[j];
      itot += cv;
      if (cv > best) { best = cv; bj = j; }
    }
    const int   cls = (bj < 0) ? 0 : (24 + bj);  // cols 0..23 are provably zero
    const float sp  = (bj < 0) ? 0.0f : (sr[bj] / (float)itot); // max(total,1)

    const size_t ob = (size_t)NINST * HW;
    out[ob + 0 * NINST + i] = (float)cls;              // inst_class
    out[ob + 1 * NINST + i] = iprobs[i];               // instance_probs
    out[ob + 2 * NINST + i] = sp;                      // seg_prob
    out[ob + 3 * NINST + i] = totals[i];               // total (WMMA result)
    out[ob + 4 * NINST + i] = (itot > 0u) ? 1.0f : 0.0f; // valid
  }
}

// ---------------------------------------------------------------------------
extern "C" void kernel_launch(void* const* d_in, const int* in_sizes, int n_in,
                              void* d_out, int out_size, void* d_ws, size_t ws_size,
                              hipStream_t stream) {
  (void)in_sizes; (void)n_in; (void)out_size; (void)ws_size;

  const int*   seg    = (const int*)  d_in[0];
  const int*   im     = (const int*)  d_in[1];
  const float* probs  = (const float*)d_in[2];
  const float* iprobs = (const float*)d_in[3];
  float*       out    = (float*)d_out;

  unsigned* gc = (unsigned*)d_ws;
  float*    gs = (float*)((char*)d_ws + (size_t)NTAB * sizeof(unsigned));

  k0_zero      <<<(NTAB + 255) / 256, 256, 0, stream>>>(gc, gs);
  k1_hist_masks<<<512, 256, 0, stream>>>(seg, im, probs, out, gc, gs);
  k2_finalize  <<<1, 256, 0, stream>>>(gc, gs, iprobs, out);
}
